// SETMIL_48026324304420
// MI455X (gfx1250) — matrix operand order repaired and checked by reading, not compiled, not run on hardware
//
#include <hip/hip_runtime.h>

// ---------------------------------------------------------------------------
// SETMIL ViT forward on MI455X (gfx1250, wave32, WMMA bf16 16x16x32)
// v3: fixed async-LDS builtin signature (AS1 v4i* / AS3 v4i*)
// ---------------------------------------------------------------------------
typedef __attribute__((ext_vector_type(16))) __bf16 v16bf;
typedef __attribute__((ext_vector_type(8)))  float  v8f;
typedef __attribute__((ext_vector_type(4)))  int    v4i;

#define EMBED   768
#define INCH    1280
#define NDEPTH  12
#define NHEADS  12
#define HDIM    64
#define DFF     3072
#define NGRID   1000
#define NBUCKET 50
#define BSZ     2
#define NTOK    4096
#define LSEQ    4097            // 1 cls + 4096 tokens
#define MROWS   (BSZ * LSEQ)    // 8194

// ---- async global->LDS DMA (CDNA5 ISA), guarded ---------
#if defined(__has_builtin)
#  if __has_builtin(__builtin_amdgcn_global_load_async_to_lds_b128)
#    define USE_ASYNC_LDS 1
#  else
#    define USE_ASYNC_LDS 0
#  endif
#  if __has_builtin(__builtin_amdgcn_s_wait_asynccnt)
#    define HAVE_WAIT_ASYNC 1
#  else
#    define HAVE_WAIT_ASYNC 0
#  endif
#else
#  define USE_ASYNC_LDS 0
#  define HAVE_WAIT_ASYNC 0
#endif

typedef __attribute__((address_space(1))) v4i GblV4;   // global int4
typedef __attribute__((address_space(3))) v4i LdsV4;   // LDS int4

// copy 16 bytes global -> LDS (async DMA when available)
__device__ __forceinline__ void async_copy16(const void* g, void* l) {
#if USE_ASYNC_LDS
    __builtin_amdgcn_global_load_async_to_lds_b128((GblV4*)g, (LdsV4*)l, 0, 0);
#else
    *(uint4*)l = *(const uint4*)g;
#endif
}
__device__ __forceinline__ void wait_async() {
#if USE_ASYNC_LDS
#  if HAVE_WAIT_ASYNC
    __builtin_amdgcn_s_wait_asynccnt(0);
#  else
    asm volatile("s_wait_asynccnt 0x0" ::: "memory");
#  endif
#endif
}

// ---- CDNA5 WMMA 16x16x32 bf16 fragment layouts (ISA 7.12.2) ----------------
// A (16x32): lane row = lane&15; elems 0..7 <-> k = hi*16+0..7,
//            elems 8..15 <-> k = hi*16+8..15  (one 32B run; k order chosen here, same as B)
// B (32x16): lane col = lane&15; elems 0..15 <-> k = hi*16+0..15 (one 32B run
//            when the tile is stored transposed [n][k] in LDS)
// C (16x16 f32): VGPR r -> row r (lanes 0-15) / r+8 (lanes 16-31), col=lane&15
union Frag {
    uint4 q[2];
    v16bf v;
};

// ---------------------------------------------------------------------------
// f32 -> bf16 conversion
// ---------------------------------------------------------------------------
__global__ void SETMIL_cvt_bf16(const float* __restrict__ s, __bf16* __restrict__ d, long n) {
    long i = (long)blockIdx.x * blockDim.x + threadIdx.x;
    if (i < n) d[i] = (__bf16)s[i];
}

// ---------------------------------------------------------------------------
// Generic bf16 GEMM: C[M,N] = act(A[M,K] @ W[K,N] + bias) (+residual)
// 64x64 tile, BK=32, 4 waves each 32x32 (2x2 WMMA). A tile async-DMA'd to LDS;
// W tile stored transposed so B-fragments are contiguous b128 LDS loads.
// ---------------------------------------------------------------------------
#define APITCH 40   // bf16 elems; 80B rows -> 16B aligned
#define BPITCH 40
__launch_bounds__(128)
__global__ void SETMIL_gemm_bf16(const __bf16* __restrict__ A, const __bf16* __restrict__ W,
                                 const float* __restrict__ bias,
                                 const float* __restrict__ residual,
                                 float* __restrict__ C, __bf16* __restrict__ Cbf,
                                 int M, int N, int K, int gelu) {
    __shared__ __bf16 As[64 * APITCH];    // [row][k]
    __shared__ __bf16 BsT[64 * BPITCH];   // [n][k]  (transposed weight tile)

    const int tid  = threadIdx.x;
    const int lane = tid & 31, wave = tid >> 5;
    const int l15  = lane & 15, hi = lane >> 4;
    const int wr   = wave >> 1, wc = wave & 1;
    const int m0   = blockIdx.y * 64, n0 = blockIdx.x * 64;

    v8f acc[2][2];
    for (int i = 0; i < 2; i++)
        for (int j = 0; j < 2; j++)
            for (int r = 0; r < 8; r++) acc[i][j][r] = 0.f;

    const int arow = tid >> 1, acol = (tid & 1) * 16;   // 64 rows x 2 halves of 16
    const int brow = tid >> 2, bcol = (tid & 3) * 16;   // 32 k-rows x 4 quarters

    for (int k0 = 0; k0 < K; k0 += 32) {
        if (k0 + 32 < K) {  // global_prefetch_b8 next tiles
            __builtin_prefetch(&A[(long)(m0 + arow) * K + k0 + 32 + acol], 0, 1);
            __builtin_prefetch(&W[(long)(k0 + 32 + brow) * N + n0 + bcol], 0, 1);
        }
        {   // A tile: row-major copy -> async DMA, 2x16B per thread
            int gr = m0 + arow;
            __bf16* dst = &As[arow * APITCH + acol];
            if (gr < M) {
                const __bf16* ap = A + (long)gr * K + k0 + acol;
                async_copy16(ap,     dst);
                async_copy16(ap + 8, dst + 8);
            } else {
                uint4 z = {0, 0, 0, 0};
                *(uint4*)dst = z; *(uint4*)(dst + 8) = z;
            }
        }
        {   // W tile: vector global load, transposed scatter into LDS
            const __bf16* bp = W + (long)(k0 + brow) * N + n0 + bcol;
            for (int u = 0; u < 2; u++) {
                uint4 d = *(const uint4*)(bp + u * 8);
                const __bf16* e8 = (const __bf16*)&d;
                for (int e = 0; e < 8; e++)
                    BsT[(bcol + u * 8 + e) * BPITCH + brow] = e8[e];
            }
        }
        wait_async();
        __syncthreads();

        Frag aF[2], bF[2];
        for (int i = 0; i < 2; i++) {
            const __bf16* p = &As[(wr * 32 + i * 16 + l15) * APITCH];
            aF[i].q[0] = *(const uint4*)(p + hi * 16);
            aF[i].q[1] = *(const uint4*)(p + hi * 16 + 8);
        }
        for (int j = 0; j < 2; j++) {
            const __bf16* p = &BsT[(wc * 32 + j * 16 + l15) * BPITCH + hi * 16];
            bF[j].q[0] = *(const uint4*)(p);
            bF[j].q[1] = *(const uint4*)(p + 8);
        }
        for (int i = 0; i < 2; i++)
            for (int j = 0; j < 2; j++)
                acc[i][j] = __builtin_amdgcn_wmma_f32_16x16x32_bf16(
                    false, aF[i].v, false, bF[j].v, (short)0, acc[i][j], false, false);
        __syncthreads();
    }

    for (int i = 0; i < 2; i++)
        for (int j = 0; j < 2; j++) {
            int gn = n0 + wc * 32 + j * 16 + l15;
            float bv = bias ? bias[gn] : 0.f;
            for (int r = 0; r < 8; r++) {
                int gm = m0 + wr * 32 + i * 16 + r + hi * 8;
                if (gm >= M) continue;
                float v = acc[i][j][r] + bv;
                if (gelu) {  // tanh-approx GELU (jax.nn.gelu default)
                    float t = 0.7978845608028654f * (v + 0.044715f * v * v * v);
                    v = 0.5f * v * (1.f + tanhf(t));
                }
                if (residual) v += residual[(long)gm * N + gn];
                C[(long)gm * N + gn] = v;
                if (Cbf) Cbf[(long)gm * N + gn] = (__bf16)v;
            }
        }
}

// ---------------------------------------------------------------------------
// LayerNorm (f32 in) -> bf16 out
// ---------------------------------------------------------------------------
__launch_bounds__(256)
__global__ void SETMIL_layernorm_bf16(const float* __restrict__ X, const float* __restrict__ w,
                                      const float* __restrict__ bp, __bf16* __restrict__ Y) {
    __shared__ float red[256];
    const int row = blockIdx.x, tid = threadIdx.x;
    const float* x = X + (long)row * EMBED;
    float s = 0.f;
    for (int c = tid; c < EMBED; c += 256) s += x[c];
    red[tid] = s; __syncthreads();
    for (int st = 128; st > 0; st >>= 1) { if (tid < st) red[tid] += red[tid + st]; __syncthreads(); }
    float mu = red[0] / EMBED; __syncthreads();
    float v = 0.f;
    for (int c = tid; c < EMBED; c += 256) { float d = x[c] - mu; v += d * d; }
    red[tid] = v; __syncthreads();
    for (int st = 128; st > 0; st >>= 1) { if (tid < st) red[tid] += red[tid + st]; __syncthreads(); }
    float rstd = rsqrtf(red[0] / EMBED + 1e-5f);
    for (int c = tid; c < EMBED; c += 256)
        Y[(long)row * EMBED + c] = (__bf16)((x[c] - mu) * rstd * w[c] + bp[c]);
}

// ---------------------------------------------------------------------------
// Patch embed epilogue + cls row + grid coords
// ---------------------------------------------------------------------------
__global__ void SETMIL_add_pos(const float* __restrict__ tmp, const float* __restrict__ coords,
                               const float* __restrict__ pos_embed, float* __restrict__ hs) {
    const int row = blockIdx.x;                 // 0..B*N-1
    const int b = row / NTOK, n = row % NTOK;
    float gx = floorf(coords[((long)b * NTOK + n) * 2 + 0] / 256.f);
    float gy = floorf(coords[((long)b * NTOK + n) * 2 + 1] / 256.f);
    int pos = (int)(gx * (float)NGRID + gy) + 1;
    const float* pe  = pos_embed + (long)pos * EMBED;
    const float* src = tmp + (long)row * EMBED;
    float* dst = hs + ((long)b * LSEQ + 1 + n) * EMBED;
    for (int c = threadIdx.x; c < EMBED; c += blockDim.x) dst[c] = src[c] + pe[c];
}

__global__ void SETMIL_cls_row(const float* __restrict__ cls_tok, const float* __restrict__ pos_embed,
                               float* __restrict__ hs) {
    const int b = blockIdx.x;
    for (int c = threadIdx.x; c < EMBED; c += blockDim.x)
        hs[(long)b * LSEQ * EMBED + c] = cls_tok[c] + pos_embed[c];
}

__global__ void SETMIL_grid_coords(const float* __restrict__ coords, float* __restrict__ gc) {
    int idx = blockIdx.x * blockDim.x + threadIdx.x;
    if (idx >= BSZ * LSEQ) return;
    int b = idx / LSEQ, l = idx % LSEQ;
    float gx = 0.f, gy = 0.f;
    if (l > 0) {
        gx = floorf(coords[((long)b * NTOK + (l - 1)) * 2 + 0] / 256.f);
        gy = floorf(coords[((long)b * NTOK + (l - 1)) * 2 + 1] / 256.f);
    }
    gc[(long)idx * 2] = gx; gc[(long)idx * 2 + 1] = gy;
}

// qkv f32 [B,L,3,H,hd] -> q(scaled)/k/v bf16 [B,H,L,hd]
__global__ void SETMIL_qkv_split(const float* __restrict__ qkv, __bf16* __restrict__ q,
                                 __bf16* __restrict__ k, __bf16* __restrict__ v) {
    long idx = (long)blockIdx.x * blockDim.x + threadIdx.x;
    const long total = (long)BSZ * NHEADS * LSEQ * HDIM;
    if (idx >= total) return;
    int d = idx % HDIM; long t = idx / HDIM;
    int l = t % LSEQ;   t /= LSEQ;
    int h = t % NHEADS; int b = (int)(t / NHEADS);
    long ro = ((long)b * LSEQ + l) * (3 * EMBED);
    int col = h * HDIM + d;
    q[idx] = (__bf16)(qkv[ro + col] * 0.125f);  // 1/sqrt(64) folded into Q
    k[idx] = (__bf16)(qkv[ro + EMBED + col]);
    v[idx] = (__bf16)(qkv[ro + 2 * EMBED + col]);
}

// attnout f32 [B,H,L,hd] -> bf16 [B*L, E]
__global__ void SETMIL_merge_heads(const float* __restrict__ ao, __bf16* __restrict__ abf) {
    long idx = (long)blockIdx.x * blockDim.x + threadIdx.x;
    const long total = (long)BSZ * LSEQ * EMBED;
    if (idx >= total) return;
    int e = idx % EMBED; long t = idx / EMBED;
    int l = t % LSEQ;    int b = (int)(t / LSEQ);
    int h = e / HDIM, d = e % HDIM;
    abf[idx] = (__bf16)ao[(((long)b * NHEADS + h) * LSEQ + l) * HDIM + d];
}

// ---------------------------------------------------------------------------
// Flash attention with fused euclidean-bucket RPE bias.
// K tile async-DMA'd row-major; V tile stored transposed [hd][key] so both
// bK and bV fragments are contiguous b128 LDS loads. P staged via LDS for
// C-layout -> A-layout conversion.
// ---------------------------------------------------------------------------
#define KPITCH 72   // bf16 elems; 144B rows -> 16B aligned
__launch_bounds__(128)
__global__ void SETMIL_flash_attn(const __bf16* __restrict__ Q, const __bf16* __restrict__ Kb,
                                  const __bf16* __restrict__ Vb, const float* __restrict__ gc,
                                  const float* __restrict__ rpe, float* __restrict__ O) {
    __shared__ __bf16 Ks[64 * KPITCH];        // [key][hd]
    __shared__ __bf16 VsT[64 * KPITCH];       // [hd][key]
    __shared__ __bf16 Ps[4][16 * KPITCH];     // per-wave P [qrow][key]
    __shared__ float  gkx[64], gky[64];
    __shared__ float  rpeS[NBUCKET];

    const int tid = threadIdx.x, lane = tid & 31, wave = tid >> 5;
    const int l15 = lane & 15, hi = lane >> 4;
    const int bh = blockIdx.y, b = bh / NHEADS, h = bh % NHEADS;
    const int qBase = blockIdx.x * 64 + wave * 16;

    const __bf16* Qh = Q + (long)bh * LSEQ * HDIM;
    const __bf16* Kh = Kb + (long)bh * LSEQ * HDIM;
    const __bf16* Vh = Vb + (long)bh * LSEQ * HDIM;
    const float*  gb = gc + (long)b * LSEQ * 2;

    if (tid < NBUCKET) rpeS[tid] = rpe[h * NBUCKET + tid];

    Frag aQ[2];
    {
        int row = qBase + l15;
        if (row < LSEQ) {
            const __bf16* qrow = Qh + (long)row * HDIM;
            for (int c = 0; c < 2; c++) {
                aQ[c].q[0] = *(const uint4*)(qrow + c * 32 + hi * 16);
                aQ[c].q[1] = *(const uint4*)(qrow + c * 32 + hi * 16 + 8);
            }
        } else {
            uint4 z = {0, 0, 0, 0};
            for (int c = 0; c < 2; c++) { aQ[c].q[0] = z; aQ[c].q[1] = z; }
        }
    }
    float gqx[8], gqy[8];
    for (int r = 0; r < 8; r++) {
        int row = qBase + r + hi * 8;
        if (row < LSEQ) { gqx[r] = gb[row * 2]; gqy[r] = gb[row * 2 + 1]; }
        else            { gqx[r] = 0.f; gqy[r] = 0.f; }
    }

    float m_i[8], l_i[8];
    v8f accO[4];
    for (int r = 0; r < 8; r++) { m_i[r] = -1e30f; l_i[r] = 0.f; }
    for (int t = 0; t < 4; t++)
        for (int r = 0; r < 8; r++) accO[t][r] = 0.f;

    const int nTiles = (LSEQ + 63) / 64;
    for (int j = 0; j < nTiles; j++) {
        {   // K tile: async DMA, V tile: vector load + transposed scatter
            int key = tid >> 1, cb = (tid & 1) * 32, kg = j * 64 + key;
            if (kg < LSEQ) {
                const __bf16* kp = Kh + (long)kg * HDIM + cb;
                __bf16* kd = &Ks[key * KPITCH + cb];
                for (int u = 0; u < 4; u++) async_copy16(kp + u * 8, kd + u * 8);
                const __bf16* vp = Vh + (long)kg * HDIM + cb;
                for (int u = 0; u < 4; u++) {
                    uint4 d = *(const uint4*)(vp + u * 8);
                    const __bf16* e8 = (const __bf16*)&d;
                    for (int e = 0; e < 8; e++)
                        VsT[(cb + u * 8 + e) * KPITCH + key] = e8[e];
                }
            } else {
                uint4 z = {0, 0, 0, 0};
                __bf16* kd = &Ks[key * KPITCH + cb];
                for (int u = 0; u < 4; u++) *(uint4*)(kd + u * 8) = z;
                for (int e = 0; e < 32; e++) VsT[(cb + e) * KPITCH + key] = (__bf16)0.f;
            }
            if (tid < 64) {
                int kg2 = j * 64 + tid;
                if (kg2 < LSEQ) { gkx[tid] = gb[kg2 * 2]; gky[tid] = gb[kg2 * 2 + 1]; }
                else            { gkx[tid] = 0.f; gky[tid] = 0.f; }
            }
        }
        wait_async();
        __syncthreads();

        v8f accS[4];
        for (int t = 0; t < 4; t++) {
            for (int r = 0; r < 8; r++) accS[t][r] = 0.f;
            for (int c = 0; c < 2; c++) {
                Frag bK;   // B = K^T chunk: contiguous hd run in Ks row
                const __bf16* p = &Ks[(t * 16 + l15) * KPITCH + c * 32 + hi * 16];
                bK.q[0] = *(const uint4*)(p);
                bK.q[1] = *(const uint4*)(p + 8);
                accS[t] = __builtin_amdgcn_wmma_f32_16x16x32_bf16(
                    false, aQ[c].v, false, bK.v, (short)0, accS[t], false, false);
            }
        }
        // RPE bias + sequence mask
        for (int t = 0; t < 4; t++) {
            int n = j * 64 + t * 16 + l15;
            float kx = gkx[t * 16 + l15], ky = gky[t * 16 + l15];
            for (int r = 0; r < 8; r++) {
                int m = qBase + r + hi * 8;
                float s;
                if (n >= LSEQ || m >= LSEQ) s = -1e30f;
                else {
                    int bucket = 0;
                    if (m != 0 && n != 0) {
                        float dx = gqx[r] - kx, dy = gqy[r] - ky;
                        int bb = (int)rintf(sqrtf(dx * dx + dy * dy));
                        bucket = bb < 0 ? 0 : (bb > NBUCKET - 1 ? NBUCKET - 1 : bb);
                    }
                    s = accS[t][r] + rpeS[bucket];
                }
                accS[t][r] = s;
            }
        }
        // online softmax (rows live across 16-lane halves)
        for (int r = 0; r < 8; r++) {
            float mx = fmaxf(fmaxf(accS[0][r], accS[1][r]), fmaxf(accS[2][r], accS[3][r]));
            for (int s = 1; s < 16; s <<= 1) mx = fmaxf(mx, __shfl_xor(mx, s, 32));
            float nm = fmaxf(m_i[r], mx);
            float sc = __expf(m_i[r] - nm);
            m_i[r] = nm; l_i[r] *= sc;
            for (int t = 0; t < 4; t++) accO[t][r] *= sc;
            float ls = 0.f;
            for (int t = 0; t < 4; t++) {
                float p = __expf(accS[t][r] - nm);
                accS[t][r] = p; ls += p;
            }
            for (int s = 1; s < 16; s <<= 1) ls += __shfl_xor(ls, s, 32);
            l_i[r] += ls;
        }
        // stage P (C-layout -> A-layout through LDS)
        for (int t = 0; t < 4; t++)
            for (int r = 0; r < 8; r++)
                Ps[wave][(r + hi * 8) * KPITCH + t * 16 + l15] = (__bf16)accS[t][r];
        __syncthreads();

        Frag aP[2];
        for (int c = 0; c < 2; c++) {
            const __bf16* p = &Ps[wave][l15 * KPITCH + c * 32];
            aP[c].q[0] = *(const uint4*)(p + hi * 16);
            aP[c].q[1] = *(const uint4*)(p + hi * 16 + 8);
        }
        for (int t = 0; t < 4; t++)
            for (int c = 0; c < 2; c++) {
                Frag bV;   // B = V chunk: contiguous key run in VsT row
                const __bf16* p = &VsT[(t * 16 + l15) * KPITCH + c * 32 + hi * 16];
                bV.q[0] = *(const uint4*)(p);
                bV.q[1] = *(const uint4*)(p + 8);
                accO[t] = __builtin_amdgcn_wmma_f32_16x16x32_bf16(
                    false, aP[c].v, false, bV.v, (short)0, accO[t], false, false);
            }
        __syncthreads();
    }

    for (int t = 0; t < 4; t++)
        for (int r = 0; r < 8; r++) {
            int row = qBase + r + hi * 8;
            if (row >= LSEQ) continue;
            float inv = (l_i[r] > 0.f) ? 1.f / l_i[r] : 0.f;
            O[((long)bh * LSEQ + row) * HDIM + t * 16 + l15] = accO[t][r] * inv;
        }
}

// ---------------------------------------------------------------------------
// Final LN on the 2 cls rows -> f32 output [B, E]
// ---------------------------------------------------------------------------
__launch_bounds__(256)
__global__ void SETMIL_final_head(const float* __restrict__ hs, const float* __restrict__ w,
                                  const float* __restrict__ bp, float* __restrict__ out) {
    __shared__ float red[256];
    const int b = blockIdx.x, tid = threadIdx.x;
    const float* x = hs + (long)b * LSEQ * EMBED;
    float s = 0.f;
    for (int c = tid; c < EMBED; c += 256) s += x[c];
    red[tid] = s; __syncthreads();
    for (int st = 128; st > 0; st >>= 1) { if (tid < st) red[tid] += red[tid + st]; __syncthreads(); }
    float mu = red[0] / EMBED; __syncthreads();
    float v = 0.f;
    for (int c = tid; c < EMBED; c += 256) { float d = x[c] - mu; v += d * d; }
    red[tid] = v; __syncthreads();
    for (int st = 128; st > 0; st >>= 1) { if (tid < st) red[tid] += red[tid + st]; __syncthreads(); }
    float rstd = rsqrtf(red[0] / EMBED + 1e-5f);
    for (int c = tid; c < EMBED; c += 256)
        out[b * EMBED + c] = (x[c] - mu) * rstd * w[c] + bp[c];
}

// ---------------------------------------------------------------------------
// Host orchestration
// ---------------------------------------------------------------------------
extern "C" void kernel_launch(void* const* d_in, const int* in_sizes, int n_in,
                              void* d_out, int out_size, void* d_ws, size_t ws_size,
                              hipStream_t stream) {
    (void)in_sizes; (void)n_in; (void)out_size; (void)ws_size;
    // setup_inputs() insertion order; 'blocks' dict leaves flattened in jax
    // pytree (alphabetical) order, each stacked [DEPTH, ...].
    const float* x        = (const float*)d_in[0];
    const float* coords   = (const float*)d_in[1];
    const float* proj_w   = (const float*)d_in[2];
    const float* proj_b   = (const float*)d_in[3];
    const float* cls_tok  = (const float*)d_in[4];
    const float* pos_emb  = (const float*)d_in[5];
    const float* attn_proj_b = (const float*)d_in[6];
    const float* attn_proj_w = (const float*)d_in[7];
    const float* fc1_b    = (const float*)d_in[8];
    const float* fc1_w    = (const float*)d_in[9];
    const float* fc2_b    = (const float*)d_in[10];
    const float* fc2_w    = (const float*)d_in[11];
    const float* ln1_b    = (const float*)d_in[12];
    const float* ln1_w    = (const float*)d_in[13];
    const float* ln2_b    = (const float*)d_in[14];
    const float* ln2_w    = (const float*)d_in[15];
    const float* qkv_w    = (const float*)d_in[16];
    const float* rpe_bias = (const float*)d_in[17];
    const float* norm_w   = (const float*)d_in[18];
    const float* norm_b   = (const float*)d_in[19];

    char* wsp = (char*)d_ws;
    size_t off = 0;
    auto alloc = [&](size_t bytes) -> void* {
        void* p = wsp + off;
        off = (off + bytes + 255) & ~(size_t)255;
        return p;
    };
    const long nXBF   = (long)BSZ * NTOK * INCH;
    const long nPW    = (long)INCH * EMBED;
    const long nQKVW  = (long)NDEPTH * EMBED * 3 * EMBED;
    const long nAPW   = (long)NDEPTH * EMBED * EMBED;
    const long nFC1W  = (long)NDEPTH * EMBED * DFF;
    const long nFC2W  = (long)NDEPTH * DFF * EMBED;
    const long nHEAD  = (long)BSZ * NHEADS * LSEQ * HDIM;

    __bf16* xbf     = (__bf16*)alloc(nXBF * 2);
    __bf16* pw_bf   = (__bf16*)alloc(nPW * 2);
    __bf16* qkvw_bf = (__bf16*)alloc(nQKVW * 2);
    __bf16* apw_bf  = (__bf16*)alloc(nAPW * 2);
    __bf16* fc1w_bf = (__bf16*)alloc(nFC1W * 2);
    __bf16* fc2w_bf = (__bf16*)alloc(nFC2W * 2);
    float*  hstream = (float*)alloc((long)MROWS * EMBED * 4);
    __bf16* hn      = (__bf16*)alloc((long)MROWS * EMBED * 2);
    float*  qkvbuf  = (float*)alloc((long)MROWS * 3 * EMBED * 4);  // also patch tmp
    __bf16* qbf     = (__bf16*)alloc(nHEAD * 2);
    __bf16* kbf     = (__bf16*)alloc(nHEAD * 2);
    __bf16* vbf     = (__bf16*)alloc(nHEAD * 2);
    float*  attnout = (float*)alloc(nHEAD * 4);
    __bf16* abf     = (__bf16*)alloc((long)MROWS * EMBED * 2);
    float*  hidden  = (float*)alloc((long)MROWS * DFF * 4);
    __bf16* hidbf   = (__bf16*)alloc((long)MROWS * DFF * 2);
    float*  gc      = (float*)alloc((long)BSZ * LSEQ * 2 * 4);

    auto cvt = [&](const float* s, __bf16* d, long n) {
        SETMIL_cvt_bf16<<<(int)((n + 255) / 256), 256, 0, stream>>>(s, d, n);
    };
    cvt(x, xbf, nXBF);
    cvt(proj_w, pw_bf, nPW);
    cvt(qkv_w, qkvw_bf, nQKVW);
    cvt(attn_proj_w, apw_bf, nAPW);
    cvt(fc1_w, fc1w_bf, nFC1W);
    cvt(fc2_w, fc2w_bf, nFC2W);

    auto gemm = [&](const __bf16* A, const __bf16* W, const float* bias, const float* res,
                    float* C, __bf16* Cbf, int M, int N, int K, int gelu) {
        dim3 g(N / 64, (M + 63) / 64);
        SETMIL_gemm_bf16<<<g, 128, 0, stream>>>(A, W, bias, res, C, Cbf, M, N, K, gelu);
    };

    // patch embed: [B*N, INCH] @ [INCH, E] + b -> qkvbuf (tmp)
    gemm(xbf, pw_bf, proj_b, nullptr, qkvbuf, nullptr, BSZ * NTOK, EMBED, INCH, 0);
    SETMIL_add_pos<<<BSZ * NTOK, 256, 0, stream>>>(qkvbuf, coords, pos_emb, hstream);
    SETMIL_cls_row<<<BSZ, 256, 0, stream>>>(cls_tok, pos_emb, hstream);
    SETMIL_grid_coords<<<(BSZ * LSEQ + 255) / 256, 256, 0, stream>>>(coords, gc);

    const long nSplit = nHEAD;
    const long nMerge = (long)MROWS * EMBED;
    for (int L = 0; L < NDEPTH; L++) {
        SETMIL_layernorm_bf16<<<MROWS, 256, 0, stream>>>(
            hstream, ln1_w + (long)L * EMBED, ln1_b + (long)L * EMBED, hn);
        gemm(hn, qkvw_bf + (long)L * EMBED * 3 * EMBED, nullptr, nullptr,
             qkvbuf, nullptr, MROWS, 3 * EMBED, EMBED, 0);
        SETMIL_qkv_split<<<(int)((nSplit + 255) / 256), 256, 0, stream>>>(qkvbuf, qbf, kbf, vbf);
        {
            dim3 g((LSEQ + 63) / 64, BSZ * NHEADS);
            SETMIL_flash_attn<<<g, 128, 0, stream>>>(
                qbf, kbf, vbf, gc, rpe_bias + (long)L * NHEADS * NBUCKET, attnout);
        }
        SETMIL_merge_heads<<<(int)((nMerge + 255) / 256), 256, 0, stream>>>(attnout, abf);
        gemm(abf, apw_bf + (long)L * EMBED * EMBED, attn_proj_b + (long)L * EMBED,
             hstream, hstream, nullptr, MROWS, EMBED, EMBED, 0);
        SETMIL_layernorm_bf16<<<MROWS, 256, 0, stream>>>(
            hstream, ln2_w + (long)L * EMBED, ln2_b + (long)L * EMBED, hn);
        gemm(hn, fc1w_bf + (long)L * EMBED * DFF, fc1_b + (long)L * DFF, nullptr,
             hidden, hidbf, MROWS, DFF, EMBED, 1);
        gemm(hidbf, fc2w_bf + (long)L * DFF * EMBED, fc2_b + (long)L * EMBED,
             hstream, hstream, nullptr, MROWS, EMBED, DFF, 0);
    }

    SETMIL_final_head<<<BSZ, 256, 0, stream>>>(hstream, norm_w, norm_b, (float*)d_out);
}
